// ContrastivePNALayer_40381282517156
// MI455X (gfx1250) — matrix operand order repaired and checked by reading, not compiled, not run on hardware
//
#include <hip/hip_runtime.h>
#include <hip/hip_bf16.h>

typedef __attribute__((ext_vector_type(16))) __bf16 v16bf;
typedef __attribute__((ext_vector_type(8)))  float  v8f;

#define FDIM   64
#define UDIM   128
#define KTOT   576          // 9 * 64
#define KSTEPS 18           // 576 / 32
#define MT     64           // node rows per block (4 sub-tiles of 16)
#define NSUB   4
#define ASTRIDE 584         // padded LDS row stride in halves (bank-conflict-free)

// ---------- helpers ----------
__device__ __forceinline__ unsigned short f2bf(float x) {
  unsigned int u = __float_as_uint(x);
  u += 0x7fffu + ((u >> 16) & 1u);          // round-to-nearest-even
  return (unsigned short)(u >> 16);
}
// order-preserving float -> uint key (0u == identity for max)
__device__ __forceinline__ unsigned int fkey(float x) {
  unsigned int b = __float_as_uint(x);
  return (b & 0x80000000u) ? ~b : (b | 0x80000000u);
}
__device__ __forceinline__ float fdecode(unsigned int k) {
  unsigned int b = (k & 0x80000000u) ? (k & 0x7fffffffu) : ~k;
  return __uint_as_float(b);
}

// ---------- 1) edge scatter: segment sum / max / count ----------
// one wave handles 32 features of one edge half; edge index is wave-uniform
__global__ void __launch_bounds__(256)
pna_scatter(const float* __restrict__ x, const long long* __restrict__ ei,
            float* __restrict__ ssum, unsigned int* __restrict__ smax,
            float* __restrict__ cnt, int E) {
  int lane   = threadIdx.x & 31;
  int waveId = threadIdx.x >> 5;
  long long gw = (long long)blockIdx.x * 8 + waveId;   // global wave
  int e = (int)(gw >> 1);
  if (e >= E) return;
  int f = ((int)(gw & 1)) * 32 + lane;
  int recv = (int)ei[2 * (long long)e];
  int send = (int)ei[2 * (long long)e + 1];
  recv = __builtin_amdgcn_readfirstlane(recv);
  send = __builtin_amdgcn_readfirstlane(send);
  float v = x[(long long)send * FDIM + f];
  atomicAdd(&ssum[(long long)recv * FDIM + f], v);
  atomicMax(&smax[(long long)recv * FDIM + f], fkey(v));
  if (f == 0) atomicAdd(&cnt[recv], 1.0f);
}

// ---------- 2) pack W (576x128 f32) into fragment-ready bf16 ----------
// frag layout: [t][ctile][lane][i], b[i] = W[t*32 + ((lane<16)?0:16) + i][ctile*16 + (lane&15)]
__global__ void pna_packW(const float* __restrict__ W, unsigned short* __restrict__ Wbf) {
  int idx = blockIdx.x * blockDim.x + threadIdx.x;
  if (idx >= KSTEPS * 8 * 32) return;
  int lane = idx & 31;
  int c    = (idx >> 5) & 7;
  int t    = idx >> 8;
  int khalf = (lane < 16) ? 0 : 16;
  int n = c * 16 + (lane & 15);
  unsigned short* dst = Wbf + (size_t)idx * 16;
#pragma unroll
  for (int i = 0; i < 16; ++i) {
    int k = t * 32 + khalf + i;
    dst[i] = f2bf(W[(size_t)k * UDIM + n]);
  }
}

// ---------- 3) feats tile in LDS + WMMA GEMM (4 M-subtiles) + relu + BN partials ----------
__global__ void __launch_bounds__(256)
pna_gemm(const float* __restrict__ ssum, const unsigned int* __restrict__ smax,
         const float* __restrict__ cnt, const unsigned short* __restrict__ Wbf,
         const float* __restrict__ bias, float* __restrict__ hbuf,
         float* __restrict__ statSum, float* __restrict__ statSq, int N) {
  __shared__ unsigned short feats[MT * ASTRIDE];   // 64 x 576 bf16, padded (~73 KB)
  int row0 = blockIdx.x * MT;
  int tid  = threadIdx.x;

  // stage A-tile: 64 rows x 64 base features -> 9 concat blocks each
  for (int p = tid; p < MT * FDIM; p += 256) {
    int r = p >> 6, f = p & 63;
    int n = row0 + r;
    float sm = 0.f, mx = 0.f, mean = 0.f, scale = 0.f;
    if (n < N) {
      float c = cnt[n];
      sm = ssum[(size_t)n * FDIM + f];
      float deg = fmaxf(c, 1.f);
      mean = sm / deg;
      mx = (c > 0.f) ? fdecode(smax[(size_t)n * FDIM + f]) : 0.f;
      scale = log10f(deg + 1.f);
    }
    unsigned short* row = &feats[r * ASTRIDE];
    unsigned short ms = f2bf(mean * scale);
    unsigned short xs = f2bf(mx * scale);
    unsigned short us = f2bf(sm * scale);
    row[0 * 64 + f] = f2bf(mean);
    row[1 * 64 + f] = ms;  row[2 * 64 + f] = ms;
    row[3 * 64 + f] = f2bf(mx);
    row[4 * 64 + f] = xs;  row[5 * 64 + f] = xs;
    row[6 * 64 + f] = f2bf(sm);
    row[7 * 64 + f] = us;  row[8 * 64 + f] = us;
  }
  __syncthreads();

  int wave = tid >> 5, lane = tid & 31;
  int m     = lane & 15;
  int kbase = (lane < 16) ? 0 : 16;  // A-frag K origin (contiguous: k = 16h + s)
  union { uint4 u[2]; v16bf v; } a, b;
  v8f acc[NSUB];
#pragma unroll
  for (int s = 0; s < NSUB; ++s) acc[s] = (v8f){};

  const uint4* bptr = (const uint4*)(Wbf + ((size_t)wave * 32 + lane) * 16);
  for (int t = 0; t < KSTEPS; ++t) {
    b.u[0] = bptr[0];
    b.u[1] = bptr[1];
    bptr += 512;               // 8 ctiles * 32 lanes * 16 halves = 8192 B
#pragma unroll
    for (int s = 0; s < NSUB; ++s) {
      const uint4* ap =
          (const uint4*)(&feats[(s * 16 + m) * ASTRIDE + t * 32 + kbase]);
      a.u[0] = ap[0];          // K = kbase .. kbase+7
      a.u[1] = ap[1];          // K = kbase+8 .. kbase+15
      acc[s] = __builtin_amdgcn_wmma_f32_16x16x32_bf16(
          false, a.v, false, b.v, (short)0, acc[s], false, false);
    }
  }

  // epilogue: bias + relu, store h, reduce per-column sum / sumsq
  int ncol  = wave * 16 + (lane & 15);
  float bn  = bias[ncol];
  int rbase = (lane < 16) ? 0 : 8;   // D layout: lanes>=16 hold rows M=8..15
  float ps = 0.f, pq = 0.f;
#pragma unroll
  for (int s = 0; s < NSUB; ++s) {
#pragma unroll
    for (int v = 0; v < 8; ++v) {
      float h = fmaxf(acc[s][v] + bn, 0.f);
      int n = row0 + s * 16 + rbase + v;
      if (n < N) {
        hbuf[(size_t)n * UDIM + ncol] = h;
        ps += h;  pq += h * h;
      }
    }
  }
  ps += __shfl_xor(ps, 16, 32);
  pq += __shfl_xor(pq, 16, 32);
  if (lane < 16) {
    atomicAdd(&statSum[ncol], ps);
    atomicAdd(&statSq[ncol],  pq);
  }
}

// ---------- 4) finalize batch statistics ----------
__global__ void pna_stats(const float* __restrict__ statSum, const float* __restrict__ statSq,
                          float* __restrict__ meanArr, float* __restrict__ invArr, float invN) {
  int c = threadIdx.x;
  float m   = statSum[c] * invN;
  float var = statSq[c] * invN - m * m;
  meanArr[c] = m;
  invArr[c]  = rsqrtf(var + 1e-3f);
}

// ---------- 5) normalize (float4 vectorized) ----------
__global__ void __launch_bounds__(256)
pna_norm(const float* __restrict__ hbuf, const float* __restrict__ meanArr,
         const float* __restrict__ invArr, const float* __restrict__ gamma,
         const float* __restrict__ beta, float* __restrict__ out, long long total4) {
  long long i = (long long)blockIdx.x * blockDim.x + threadIdx.x;
  if (i >= total4) return;
  float4 h = ((const float4*)hbuf)[i];
  int c0 = (int)((i * 4) & (UDIM - 1));
  float4 o;
  o.x = gamma[c0 + 0] * (h.x - meanArr[c0 + 0]) * invArr[c0 + 0] + beta[c0 + 0];
  o.y = gamma[c0 + 1] * (h.y - meanArr[c0 + 1]) * invArr[c0 + 1] + beta[c0 + 1];
  o.z = gamma[c0 + 2] * (h.z - meanArr[c0 + 2]) * invArr[c0 + 2] + beta[c0 + 2];
  o.w = gamma[c0 + 3] * (h.w - meanArr[c0 + 3]) * invArr[c0 + 3] + beta[c0 + 3];
  ((float4*)out)[i] = o;
}

extern "C" void kernel_launch(void* const* d_in, const int* in_sizes, int n_in,
                              void* d_out, int out_size, void* d_ws, size_t ws_size,
                              hipStream_t stream) {
  const float*     x     = (const float*)d_in[0];
  const long long* ei    = (const long long*)d_in[1];  // int64 per reference
  const float*     W     = (const float*)d_in[2];
  const float*     bias  = (const float*)d_in[3];
  const float*     gamma = (const float*)d_in[4];
  const float*     beta  = (const float*)d_in[5];
  float* out = (float*)d_out;

  int N = in_sizes[0] / FDIM;
  int E = in_sizes[1] / 2;

  // workspace carve-out (256B aligned)
  char* ws = (char*)d_ws;
  size_t off = 0;
  auto carve = [&](size_t bytes) -> void* {
    void* p = ws + off;
    off = (off + bytes + 255) & ~(size_t)255;
    return p;
  };
  float*          ssum    = (float*)carve((size_t)N * FDIM * 4);
  unsigned int*   smax    = (unsigned int*)carve((size_t)N * FDIM * 4);
  float*          cnt     = (float*)carve((size_t)N * 4);
  float*          statSum = (float*)carve(UDIM * 4);
  float*          statSq  = (float*)carve(UDIM * 4);
  size_t zeroBytes = off;                          // everything above needs zero-init
  unsigned short* Wbf     = (unsigned short*)carve((size_t)KSTEPS * 8 * 32 * 16 * 2);
  float*          meanArr = (float*)carve(UDIM * 4);
  float*          invArr  = (float*)carve(UDIM * 4);
  float*          hbuf    = (float*)carve((size_t)N * UDIM * 4);

  hipMemsetAsync(ws, 0, zeroBytes, stream);

  long long waves = (long long)E * 2;              // 2 waves per edge (64 feats)
  int scatterBlocks = (int)((waves + 7) / 8);
  pna_scatter<<<scatterBlocks, 256, 0, stream>>>(x, ei, ssum, smax, cnt, E);

  pna_packW<<<(KSTEPS * 8 * 32 + 255) / 256, 256, 0, stream>>>(W, Wbf);

  int gemmBlocks = (N + MT - 1) / MT;
  pna_gemm<<<gemmBlocks, 256, 0, stream>>>(ssum, smax, cnt, Wbf, bias,
                                           hbuf, statSum, statSq, N);

  pna_stats<<<1, UDIM, 0, stream>>>(statSum, statSq, meanArr, invArr, 1.0f / (float)N);

  long long total4 = (long long)N * UDIM / 4;
  int normBlocks = (int)((total4 + 255) / 256);
  pna_norm<<<normBlocks, 256, 0, stream>>>(hbuf, meanArr, invArr, gamma, beta, out, total4);
}
